// LSTM_45114336477965
// MI455X (gfx1250) — matrix operand order repaired and checked by reading, not compiled, not run on hardware
//
#include <hip/hip_runtime.h>

// ---------------------------------------------------------------------------
// 2-layer LSTM for MI455X (gfx1250), wave32 + WMMA bf16.
//   B=32, S=512, I=H=1024, L=2, 4H=4096
// Phase 1 per layer: gx[s,b,n] = sum_k xb[s,b,k]*w_ih[n,k] + (b_ih+b_hh)[n]
//   -> big bf16 WMMA GEMM, f32 accumulate, ping-pong pipelined fragment loads
//      (no register rotation copies), non-temporal gx stores.
// Phase 2 per layer: persistent recurrence kernel, 128 WGs, w_hh slice in LDS,
//   c in registers, grid barrier per timestep, h ping-pong in global bf16.
// ---------------------------------------------------------------------------

#define BB   32
#define SS   512
#define HH   1024
#define GG   4096          // 4*H
#define NWG  128           // workgroups in recurrence kernel
#define JPW  8             // hidden units per WG (HH / NWG)

typedef __attribute__((ext_vector_type(16))) __bf16   v16bf;
typedef __attribute__((ext_vector_type(8)))  float    v8f;
typedef __attribute__((ext_vector_type(4)))  unsigned u4e;     // 16B vector

union FragBF { u4e q[2]; v16bf v; };

__device__ __forceinline__ unsigned short f2bf(float f) {
    unsigned u = __builtin_bit_cast(unsigned, f);
    u = u + 0x7FFFu + ((u >> 16) & 1u);          // round-to-nearest-even
    return (unsigned short)(u >> 16);
}

__device__ __forceinline__ float sigmoidf_(float x) {
    return 1.0f / (1.0f + __expf(-x));
}
__device__ __forceinline__ float tanhf_(float x) {
    x = fminf(fmaxf(x, -15.0f), 15.0f);
    float e = __expf(-2.0f * x);
    return (1.0f - e) / (1.0f + e);
}

// A fragment: 16x32 bf16, A row-major [M,K]. Lanes 0-15 row M=lane,
// K = k0..k0+15 ; lanes 16-31: K = k0+16..k0+31 (matches B fragment).
__device__ __forceinline__ v16bf load_a_frag_g(const unsigned short* A, int lda,
                                               int mBase, int k0, int lane) {
    int m    = mBase + (lane & 15);
    int half = lane >> 4;
    const unsigned short* p = A + (long)m * lda + k0 + half * 16;
    FragBF fr;
    fr.q[0] = *(const u4e*)p;          // K = k0 + half*16 + [0..7]
    fr.q[1] = *(const u4e*)(p + 8);    // K = k0 + half*16 + 8 + [0..7]
    return fr.v;
}

// B fragment: 32x16 bf16, B[k][n] = W[n][k] with W row-major [N,K].
// Lanes 0-15: column n = lane, K = k0..k0+15 ; lanes 16-31: K = k0+16..k0+31.
__device__ __forceinline__ v16bf load_b_frag_g(const unsigned short* W, int ldb,
                                               int nBase, int k0, int lane) {
    int n    = nBase + (lane & 15);
    int half = lane >> 4;
    const unsigned short* p = W + (long)n * ldb + k0 + half * 16;
    FragBF fr;
    fr.q[0] = *(const u4e*)p;
    fr.q[1] = *(const u4e*)(p + 8);
    return fr.v;
}

// ---------------------------------------------------------------------------
// f32 -> bf16 bulk convert
__global__ void k_cvt_bf16(const float* __restrict__ src,
                           unsigned short* __restrict__ dst, int n) {
    for (int i = blockIdx.x * blockDim.x + threadIdx.x; i < n;
         i += gridDim.x * blockDim.x)
        dst[i] = f2bf(src[i]);
}

// combined bias = b_ih + b_hh (f32)
__global__ void k_bias(const float* __restrict__ bih,
                       const float* __restrict__ bhh,
                       float* __restrict__ bias) {
    int i = blockIdx.x * blockDim.x + threadIdx.x;
    if (i < GG) bias[i] = bih[i] + bhh[i];
}

// [B,S,1024] f32 -> [S,B,1024] bf16
__global__ void k_x2bf(const float* __restrict__ src,
                       unsigned short* __restrict__ dst) {
    long total = (long)BB * SS * 1024;
    for (long i = (long)blockIdx.x * blockDim.x + threadIdx.x; i < total;
         i += (long)gridDim.x * blockDim.x) {
        int  k = (int)(i & 1023);
        long t = i >> 10;
        int  s = (int)(t & (SS - 1));
        int  b = (int)(t >> 9);
        dst[((long)s * BB + b) * 1024 + k] = f2bf(src[i]);
    }
}

// zero h ping-pong buffer + sync counters
__global__ void k_init(unsigned short* __restrict__ hbuf,
                       unsigned* __restrict__ sync) {
    int i = blockIdx.x * blockDim.x + threadIdx.x;
    if (i < 2 * BB * HH) hbuf[i] = 0;
    if (i < 32) sync[i] = 0;
}

// ---------------------------------------------------------------------------
// gx = A(bf16 [M=S*B,K=1024]) @ W(bf16 [N=4096,K=1024])^T + bias, f32 out.
// Grid: (16, 512) blocks of 256 threads; each wave: 1 M-tile x 4 N-tiles.
// Ping-pong pipelined: stage Q loads while P computes, and vice versa.
__global__ __launch_bounds__(256) void k_gemm_gx(
        const unsigned short* __restrict__ A,
        const unsigned short* __restrict__ W,
        const float* __restrict__ bias,
        float* __restrict__ gx) {
    const int K = 1024;
    int lane = threadIdx.x & 31;
    int w    = threadIdx.x >> 5;                 // 0..7
    int mi   = w & 1;
    int nq   = w >> 1;                           // 0..3
    int mBase  = blockIdx.y * 32 + mi * 16;
    int nBase0 = blockIdx.x * 256 + nq * 64;

    v8f acc[4] = {v8f{}, v8f{}, v8f{}, v8f{}};

    v16bf aP, aQ, bP[4], bQ[4];
    aP = load_a_frag_g(A, K, mBase, 0, lane);
#pragma unroll
    for (int t = 0; t < 4; ++t)
        bP[t] = load_b_frag_g(W, K, nBase0 + t * 16, 0, lane);

    for (int k0 = 0; k0 < K; k0 += 64) {
        // stage Q <- k0+32 while computing on P (k0)
        aQ = load_a_frag_g(A, K, mBase, k0 + 32, lane);
#pragma unroll
        for (int t = 0; t < 4; ++t)
            bQ[t] = load_b_frag_g(W, K, nBase0 + t * 16, k0 + 32, lane);
        __builtin_prefetch(A + (long)(mBase + (lane & 15)) * K + k0 + 128, 0, 1);
#pragma unroll
        for (int t = 0; t < 4; ++t)
            acc[t] = __builtin_amdgcn_wmma_f32_16x16x32_bf16(
                false, aP, false, bP[t], (short)0, acc[t], false, false);

        // stage P <- k0+64 (wrap on last iteration: benign) while computing Q
        int k2 = (k0 + 64) & (K - 1);
        aP = load_a_frag_g(A, K, mBase, k2, lane);
#pragma unroll
        for (int t = 0; t < 4; ++t)
            bP[t] = load_b_frag_g(W, K, nBase0 + t * 16, k2, lane);
#pragma unroll
        for (int t = 0; t < 4; ++t)
            acc[t] = __builtin_amdgcn_wmma_f32_16x16x32_bf16(
                false, aQ, false, bQ[t], (short)0, acc[t], false, false);
    }

    int half = lane >> 4, nl = lane & 15;
#pragma unroll
    for (int t = 0; t < 4; ++t) {
        int   n  = nBase0 + t * 16 + nl;
        float bv = bias[n];
#pragma unroll
        for (int r = 0; r < 8; ++r) {
            int m = mBase + half * 8 + r;
            // gx is streamed (read once later): non-temporal store
            __builtin_nontemporal_store(acc[t][r] + bv, &gx[(long)m * GG + n]);
        }
    }
}

// ---------------------------------------------------------------------------
// Persistent recurrence. 128 WGs x 128 threads (4 waves). WG owns 8 hidden
// units => 32 gate rows of w_hh staged in LDS (64KB bf16). c in registers.
__global__ __launch_bounds__(128) void k_lstm(
        const float* __restrict__ gx,            // [S*B, 4096]
        const unsigned short* __restrict__ whh,  // bf16 [4096, 1024]
        unsigned short* __restrict__ hbuf,       // bf16 [2][B][H]
        unsigned* __restrict__ sync,
        float* __restrict__ outHS,               // [B, S, H]
        float* __restrict__ hn,                  // [B, H]
        float* __restrict__ cn) {                // [B, H]
    extern __shared__ unsigned short smem[];
    unsigned short* whh_s = smem;                        // [32][1024] bf16
    float* gates = (float*)(smem + 32 * 1024);           // [32 batch][32 rows]

    int tid   = threadIdx.x;
    int jBase = blockIdx.x * JPW;

    // stage this WG's 32 gate rows of w_hh into LDS (row r = g*8 + jj)
    for (int idx = tid; idx < 32 * 128; idx += 128) {
        int r  = idx >> 7;                               // 0..31
        int c8 = (idx & 127) * 8;                        // 8 bf16 per copy
        int gr = (r >> 3) * HH + jBase + (r & 7);        // global gate row
        *(u4e*)&whh_s[r * 1024 + c8] = *(const u4e*)&whh[(long)gr * HH + c8];
    }
    __syncthreads();

    int lane = tid & 31, w = tid >> 5;                   // 4 waves
    int mBase = (w & 1) * 16;                            // batch tile
    int nBase = (w >> 1) * 16;                           // gate-row tile
    int nlB   = nBase + (lane & 15);                     // LDS B row for frags
    int khB   = (lane >> 4) * 16;
    int b0 = tid >> 3, jj = tid & 7;                     // cell pairs
    int b1 = b0 + 16;
    float c0 = 0.0f, c1 = 0.0f;

    unsigned* counter = sync;
    unsigned* flag    = sync + 16;

    for (int s = 0; s < SS; ++s) {
        const unsigned short* hp = hbuf + (s & 1) * BB * HH;
        unsigned short* hnw = hbuf + ((s & 1) ^ 1) * BB * HH;

        // gates_tile[32b x 32r] = h_prev @ whh_sliceᵀ (K=1024), ping-pong
        v8f acc = {};
        v16bf aP, aQ;
        FragBF fbP, fbQ;
        aP = load_a_frag_g(hp, HH, mBase, 0, lane);
        fbP.q[0] = *(const u4e*)&whh_s[nlB * 1024 + khB];
        fbP.q[1] = *(const u4e*)&whh_s[nlB * 1024 + khB + 8];
        for (int k0 = 0; k0 < HH; k0 += 64) {
            aQ = load_a_frag_g(hp, HH, mBase, k0 + 32, lane);
            fbQ.q[0] = *(const u4e*)&whh_s[nlB * 1024 + k0 + 32 + khB];
            fbQ.q[1] = *(const u4e*)&whh_s[nlB * 1024 + k0 + 32 + khB + 8];
            acc = __builtin_amdgcn_wmma_f32_16x16x32_bf16(
                false, aP, false, fbP.v, (short)0, acc, false, false);

            int k2 = (k0 + 64) & (HH - 1);
            aP = load_a_frag_g(hp, HH, mBase, k2, lane);
            fbP.q[0] = *(const u4e*)&whh_s[nlB * 1024 + k2 + khB];
            fbP.q[1] = *(const u4e*)&whh_s[nlB * 1024 + k2 + khB + 8];
            acc = __builtin_amdgcn_wmma_f32_16x16x32_bf16(
                false, aQ, false, fbQ.v, (short)0, acc, false, false);
        }
        {
            int half = lane >> 4, nl = lane & 15;
#pragma unroll
            for (int r = 0; r < 8; ++r)
                gates[(mBase + half * 8 + r) * 32 + nBase + nl] = acc[r];
        }
        __syncthreads();

        // elementwise cell update: this thread owns (b0,jj) and (b1,jj)
        {
            int col = jBase + jj;
            const float* gp0 = gx + ((long)(s * BB + b0)) * GG + col;
            const float* gp1 = gx + ((long)(s * BB + b1)) * GG + col;

            // gx is read exactly once: non-temporal loads
            float i0 = sigmoidf_(gates[b0 * 32 + jj]      + __builtin_nontemporal_load(gp0));
            float f0 = sigmoidf_(gates[b0 * 32 + 8 + jj]  + __builtin_nontemporal_load(gp0 + 1024));
            float g0 = tanhf_   (gates[b0 * 32 + 16 + jj] + __builtin_nontemporal_load(gp0 + 2048));
            float o0 = sigmoidf_(gates[b0 * 32 + 24 + jj] + __builtin_nontemporal_load(gp0 + 3072));
            c0 = f0 * c0 + i0 * g0;
            float h0 = o0 * tanhf_(c0);

            float i1 = sigmoidf_(gates[b1 * 32 + jj]      + __builtin_nontemporal_load(gp1));
            float f1 = sigmoidf_(gates[b1 * 32 + 8 + jj]  + __builtin_nontemporal_load(gp1 + 1024));
            float g1 = tanhf_   (gates[b1 * 32 + 16 + jj] + __builtin_nontemporal_load(gp1 + 2048));
            float o1 = sigmoidf_(gates[b1 * 32 + 24 + jj] + __builtin_nontemporal_load(gp1 + 3072));
            c1 = f1 * c1 + i1 * g1;
            float h1 = o1 * tanhf_(c1);

            hnw[b0 * HH + col] = f2bf(h0);
            hnw[b1 * HH + col] = f2bf(h1);
            outHS[(long)b0 * (SS * HH) + (long)s * HH + col] = h0;
            outHS[(long)b1 * (SS * HH) + (long)s * HH + col] = h1;
            if (s == SS - 1) {
                hn[b0 * HH + col] = h0;  hn[b1 * HH + col] = h1;
                cn[b0 * HH + col] = c0;  cn[b1 * HH + col] = c1;
            }
        }
        __syncthreads();

        // grid-wide barrier (ticket / generation scheme, s_sleep backoff)
        if (tid == 0) {
            __threadfence();
            unsigned t   = __atomic_fetch_add(counter, 1u, __ATOMIC_ACQ_REL);
            unsigned gen = t / NWG + 1;
            if (t % NWG == NWG - 1) {
                __atomic_store_n(flag, gen, __ATOMIC_RELEASE);
            } else {
                while (__atomic_load_n(flag, __ATOMIC_ACQUIRE) < gen)
                    __builtin_amdgcn_s_sleep(2);
            }
        }
        __syncthreads();
    }
}

// ---------------------------------------------------------------------------
extern "C" void kernel_launch(void* const* d_in, const int* in_sizes, int n_in,
                              void* d_out, int out_size, void* d_ws,
                              size_t ws_size, hipStream_t stream) {
    const float* x    = (const float*)d_in[0];
    const float* w_ih = (const float*)d_in[1];   // [L, 4H, I]
    const float* w_hh = (const float*)d_in[2];   // [L, 4H, H]
    const float* b_ih = (const float*)d_in[3];   // [L, 4H]
    const float* b_hh = (const float*)d_in[4];   // [L, 4H]

    float* out = (float*)d_out;                  // [B,S,H] ++ hn[2,B,H] ++ cn[2,B,H]
    float* hn  = out + (long)BB * SS * HH;
    float* cn  = hn + 2L * BB * HH;

    char* ws = (char*)d_ws;
    size_t off = 0;
    float*          gx   = (float*)(ws + off);          off += (size_t)SS * BB * GG * 4;   // 256MB
    unsigned short* xb   = (unsigned short*)(ws + off); off += (size_t)SS * BB * 1024 * 2; // 32MB
    unsigned short* wihb = (unsigned short*)(ws + off); off += (size_t)GG * 1024 * 2;      // 8MB
    unsigned short* whhb = (unsigned short*)(ws + off); off += (size_t)GG * 1024 * 2;      // 8MB
    float*          bias = (float*)(ws + off);          off += (size_t)GG * 4;
    unsigned short* hbuf = (unsigned short*)(ws + off); off += (size_t)2 * BB * HH * 2;
    unsigned*       sync = (unsigned*)(ws + off);       off += 128;

    const size_t lds_bytes = 32 * 1024 * 2 + 32 * 32 * 4;  // 68KB (320KB/WGP avail)

    for (int l = 0; l < 2; ++l) {
        k_cvt_bf16<<<1024, 256, 0, stream>>>(w_ih + (long)l * GG * 1024, wihb, GG * 1024);
        k_cvt_bf16<<<1024, 256, 0, stream>>>(w_hh + (long)l * GG * 1024, whhb, GG * 1024);
        k_bias<<<16, 256, 0, stream>>>(b_ih + l * GG, b_hh + l * GG, bias);
        // layer 0 input = x; layer 1 input = layer-0 hs (temporarily in d_out)
        k_x2bf<<<2048, 256, 0, stream>>>(l == 0 ? x : out, xb);
        k_gemm_gx<<<dim3(16, 512), 256, 0, stream>>>(xb, wihb, bias, gx);
        k_init<<<256, 256, 0, stream>>>(hbuf, sync);
        k_lstm<<<NWG, 128, lds_bytes, stream>>>(gx, whhb, hbuf, sync,
                                                out, hn + (long)l * BB * HH,
                                                cn + (long)l * BB * HH);
    }
}
